// pBLSTMlayer_57638461112588
// MI455X (gfx1250) — compile-verified, matrix operand order checked by reading, not run on hardware
//
#include <hip/hip_runtime.h>
#include <hip/hip_bf16.h>

// Bidirectional LSTM, B=64 T=800 D=512 H=256.
// Phase 1: pack weights to bf16, transposed (N-major, K-contiguous) in a
//          "gate-packed" column order so each scan WG's 256 gate columns are
//          contiguous.  Zero h double-buffers + barrier counters.
// Phase 2: xproj[t,b,:] = x[b,t,:] @ Wx + bias  via v_wmma_f32_16x16x32_bf16,
//          fp32 accumulate, stored bf16.  Fully parallel over (t, nblk, dir).
// Phase 3: persistent scan: 8 workgroups (4 per direction, 64 hidden units
//          each).  Per step: h@Wh via WMMA (Wh slice streamed from L2),
//          gate combine with exp-based sigmoid/tanh, c kept in LDS,
//          h exchanged through global bf16 double buffer with an atomic
//          arrive/spin barrier between the 4 WGs of a direction.

#define Bn 64
#define Tn 800
#define Dn 512
#define Hn 256
#define Gn 1024  // 4H

typedef __attribute__((ext_vector_type(8)))  float  v8f;
typedef __attribute__((ext_vector_type(8)))  __bf16 v8bf;
typedef __attribute__((ext_vector_type(16))) __bf16 v16bf;

union ABu  { v16bf v; v8bf h[2]; };
union ACCu { v8f   v; float f[8]; };

__device__ __forceinline__ v8f zero8() {
  v8f z = {0.f,0.f,0.f,0.f,0.f,0.f,0.f,0.f};
  return z;
}

// Build a 16x32 bf16 WMMA A/B operand for one lane.
// base points at the start of this lane's row (A) / column (B), K-contiguous.
// Per ISA 7.12.2: lanes 0-15 hold K in {K0..K0+7, K0+16..K0+23},
// lanes 16-31 hold K in {K0+8..K0+15, K0+24..K0+31}.
__device__ __forceinline__ v16bf load_ab(const __bf16* base, int K0, int hi) {
  ABu u;
  u.h[0] = *(const v8bf*)(base + K0 + hi * 8);
  u.h[1] = *(const v8bf*)(base + K0 + 16 + hi * 8);
  return u.v;
}

__device__ __forceinline__ float sigm_f(float x) {
  return 1.0f / (1.0f + __expf(-x));
}
__device__ __forceinline__ float tanh_f(float x) {
  // 1 - 2/(e^{2x}+1): saturates correctly for |x| large.
  float e = __expf(2.0f * x);
  return 1.0f - 2.0f / (e + 1.0f);
}

// packed column p = slice*256 + gate*64 + jj  ->  original gate column
__device__ __forceinline__ int orig_col(int p) {
  return ((p >> 6) & 3) * Hn + (p >> 8) * 64 + (p & 63);
}

// ---------------------------------------------------------------------------
// Phase 1: pack weights / biases, zero h buffers and counters.
// ---------------------------------------------------------------------------
__global__ void __launch_bounds__(256)
pack_kernel(const float* __restrict__ Wxf, const float* __restrict__ Whf,
            const float* __restrict__ bfo,
            const float* __restrict__ Wxb, const float* __restrict__ Whb,
            const float* __restrict__ bbo,
            __bf16* __restrict__ WxT,   // [2][1024][512]
            __bf16* __restrict__ WhT,   // [2][1024][256]
            float*  __restrict__ biasP, // [2][1024]
            __bf16* __restrict__ hbuf,  // [2][2][64][256]
            int*    __restrict__ cnt)   // [2]
{
  const int idx = blockIdx.x * 256 + threadIdx.x;  // grid covers 2*1024*512
  const int NX = 1024 * Dn;
  if (idx < 2 * NX) {
    int d = idx / NX, r = idx % NX;
    int p = r >> 9, k = r & 511;
    const float* W = d ? Wxb : Wxf;
    WxT[idx] = (__bf16)W[(size_t)k * Gn + orig_col(p)];
  }
  const int NH = 1024 * Hn;
  if (idx < 2 * NH) {
    int d = idx / NH, r = idx % NH;
    int p = r >> 8, k = r & 255;
    const float* W = d ? Whb : Whf;
    WhT[idx] = (__bf16)W[(size_t)k * Gn + orig_col(p)];
  }
  if (idx < 2 * 1024) {
    int d = idx >> 10, p = idx & 1023;
    biasP[idx] = (d ? bbo : bfo)[orig_col(p)];
  }
  if (idx < 2 * 2 * Bn * Hn) hbuf[idx] = (__bf16)0.0f;
  if (idx < 2) cnt[idx] = 0;
}

// ---------------------------------------------------------------------------
// Phase 2: xproj GEMM.  grid = (800, 8): x = t, y = dir*4 + nblk.
// Each WG: 64 rows (all batches at time t) x 256 packed gate cols.
// 256 threads = 8 waves; wave w owns 32 cols (2 N-tiles), 4 M-tiles, K=512.
// ---------------------------------------------------------------------------
__global__ void __launch_bounds__(256)
xproj_kernel(const float* __restrict__ x,     // [64][800][512]
             const __bf16* __restrict__ WxT,  // [2][1024][512]
             const float*  __restrict__ biasP,// [2][1024]
             __bf16* __restrict__ xp)         // [2][800][64][1024]
{
  extern __shared__ char smem[];
  __bf16* ash = (__bf16*)smem;  // [64][512] bf16 = 64KB

  const int tid = threadIdx.x;
  const int wave = tid >> 5, lane = tid & 31;
  const int lane15 = lane & 15, hi = lane >> 4;
  const int t = blockIdx.x;
  const int dir = blockIdx.y >> 2, nblk = blockIdx.y & 3;

  // Stage A: x[b][t][:] for b=0..63, converted to bf16.
  for (int i = tid; i < Bn * Dn; i += 256) {
    int b = i >> 9, k = i & 511;
    ash[i] = (__bf16)x[((size_t)b * Tn + t) * Dn + k];
  }
  __syncthreads();

  const __bf16* Wd = WxT + (size_t)dir * 1024 * Dn;
  const int nbase = nblk * 256 + wave * 32;

  v8f acc[2][4];
#pragma unroll
  for (int nt = 0; nt < 2; ++nt)
#pragma unroll
    for (int mt = 0; mt < 4; ++mt) acc[nt][mt] = zero8();

#pragma unroll 4
  for (int kc = 0; kc < 16; ++kc) {
    const int K0 = kc * 32;
    v16bf a[4], bm[2];
#pragma unroll
    for (int mt = 0; mt < 4; ++mt)
      a[mt] = load_ab(ash + (mt * 16 + lane15) * Dn, K0, hi);
#pragma unroll
    for (int nt = 0; nt < 2; ++nt)
      bm[nt] = load_ab(Wd + (size_t)(nbase + nt * 16 + lane15) * Dn, K0, hi);
#pragma unroll
    for (int nt = 0; nt < 2; ++nt)
#pragma unroll
      for (int mt = 0; mt < 4; ++mt)
        acc[nt][mt] = __builtin_amdgcn_wmma_f32_16x16x32_bf16(
            false, a[mt], false, bm[nt], (short)0, acc[nt][mt], false, false);
  }

  const float* bp = biasP + dir * 1024;
  __bf16* xpd = xp + (size_t)dir * Tn * Bn * Gn;
#pragma unroll
  for (int nt = 0; nt < 2; ++nt) {
    const int p = nbase + nt * 16 + lane15;
    const float bias = bp[p];
#pragma unroll
    for (int mt = 0; mt < 4; ++mt) {
      ACCu u; u.v = acc[nt][mt];
#pragma unroll
      for (int r = 0; r < 8; ++r) {
        const int M = mt * 16 + r + hi * 8;  // batch index
        xpd[((size_t)t * Bn + M) * Gn + p] = (__bf16)(u.f[r] + bias);
      }
    }
  }
}

// ---------------------------------------------------------------------------
// Phase 3: persistent recurrent scan.  8 WGs: blockIdx 0-3 = fwd slices,
// 4-7 = bwd slices.  Each WG owns 64 hidden units (256 packed gate cols).
// LDS: h stage 32KB + gate exchange 64KB + c state 16KB = 112KB.
// Wh slice (128KB bf16) streamed from L2 every step.
// ---------------------------------------------------------------------------
__global__ void __launch_bounds__(256)
lstm_scan_kernel(const __bf16* __restrict__ xp,   // [2][800][64][1024]
                 const __bf16* __restrict__ WhT,  // [2][1024][256]
                 __bf16* __restrict__ hbuf,       // [2][2][64][256]
                 int* __restrict__ cnt,           // [2]
                 float* __restrict__ out)         // [64][800][512]
{
  extern __shared__ char smem[];
  __bf16* hsh = (__bf16*)smem;                     // [64][256] bf16 = 32KB
  float*  gsh = (float*)(smem + 32768);            // [64][256] f32  = 64KB
  float*  csh = (float*)(smem + 32768 + 65536);    // [64][64]  f32  = 16KB

  const int tid = threadIdx.x;
  const int wave = tid >> 5, lane = tid & 31;
  const int lane15 = lane & 15, hi = lane >> 4;
  const int dir = blockIdx.x >> 2, slice = blockIdx.x & 3;

  const __bf16* Whd = WhT + ((size_t)dir * 1024 + slice * 256) * Hn;
  const __bf16* xpd = xp + (size_t)dir * Tn * Bn * Gn;
  int* mycnt = cnt + dir;

  for (int i = tid; i < Bn * 64; i += 256) csh[i] = 0.0f;
  __syncthreads();

  for (int s = 0; s < Tn; ++s) {
    const int t = dir ? (Tn - 1 - s) : s;

    // Stage h_prev (full H) from global double buffer into LDS.
    {
      const v8bf* src =
          (const v8bf*)(hbuf + ((size_t)(dir * 2 + (s & 1))) * Bn * Hn);
      v8bf* dst = (v8bf*)hsh;
      for (int i = tid; i < Bn * Hn / 8; i += 256) dst[i] = src[i];
    }
    __syncthreads();

    // GEMM: h(64x256) @ Wh_slice(256x256) -> gate pre-activations.
    v8f acc[2][4];
#pragma unroll
    for (int nt = 0; nt < 2; ++nt)
#pragma unroll
      for (int mt = 0; mt < 4; ++mt) acc[nt][mt] = zero8();

#pragma unroll
    for (int kc = 0; kc < 8; ++kc) {
      const int K0 = kc * 32;
      v16bf a[4], bm[2];
#pragma unroll
      for (int mt = 0; mt < 4; ++mt)
        a[mt] = load_ab(hsh + (mt * 16 + lane15) * Hn, K0, hi);
#pragma unroll
      for (int nt = 0; nt < 2; ++nt)
        bm[nt] = load_ab(Whd + (size_t)(wave * 32 + nt * 16 + lane15) * Hn,
                         K0, hi);
#pragma unroll
      for (int nt = 0; nt < 2; ++nt)
#pragma unroll
        for (int mt = 0; mt < 4; ++mt)
          acc[nt][mt] = __builtin_amdgcn_wmma_f32_16x16x32_bf16(
              false, a[mt], false, bm[nt], (short)0, acc[nt][mt], false, false);
    }

    // Scatter D tiles to LDS (f32 16x16 C/D layout, ISA 7.12.2).
#pragma unroll
    for (int nt = 0; nt < 2; ++nt) {
      const int ncol = wave * 32 + nt * 16 + lane15;
#pragma unroll
      for (int mt = 0; mt < 4; ++mt) {
        ACCu u; u.v = acc[nt][mt];
#pragma unroll
        for (int r = 0; r < 8; ++r) {
          const int M = mt * 16 + r + hi * 8;
          gsh[M * 256 + ncol] = u.f[r];
        }
      }
    }
    __syncthreads();

    // Gate combine: 64 batches x 64 hidden units owned by this WG.
    for (int idx = tid; idx < Bn * 64; idx += 256) {
      const int b = idx >> 6, jj = idx & 63;
      const size_t xbase = ((size_t)t * Bn + b) * Gn + slice * 256 + jj;
      const float gi = gsh[b * 256 + 0 * 64 + jj] + (float)xpd[xbase + 0 * 64];
      const float gf = gsh[b * 256 + 1 * 64 + jj] + (float)xpd[xbase + 1 * 64];
      const float gg = gsh[b * 256 + 2 * 64 + jj] + (float)xpd[xbase + 2 * 64];
      const float go = gsh[b * 256 + 3 * 64 + jj] + (float)xpd[xbase + 3 * 64];
      const float c = sigm_f(gf) * csh[idx] + sigm_f(gi) * tanh_f(gg);
      const float h = sigm_f(go) * tanh_f(c);
      csh[idx] = c;
      const int j = slice * 64 + jj;
      out[((size_t)b * Tn + t) * (2 * Hn) + dir * Hn + j] = h;
      hbuf[((size_t)(dir * 2 + ((s + 1) & 1))) * Bn * Hn + b * Hn + j] =
          (__bf16)h;
    }

    // Cross-WG barrier among the 4 WGs of this direction.
    __threadfence();
    __syncthreads();
    if (tid == 0) {
      __hip_atomic_fetch_add(mycnt, 1, __ATOMIC_ACQ_REL,
                             __HIP_MEMORY_SCOPE_AGENT);
      const int target = 4 * (s + 1);
      while (__hip_atomic_load(mycnt, __ATOMIC_ACQUIRE,
                               __HIP_MEMORY_SCOPE_AGENT) < target)
        __builtin_amdgcn_s_sleep(1);
    }
    __syncthreads();
  }
}

// ---------------------------------------------------------------------------
// Host launcher.
// Workspace layout (bytes):
//   [0)          WxT   2*1024*512 bf16  = 2,097,152
//   [2,097,152)  WhT   2*1024*256 bf16  = 1,048,576
//   [3,145,728)  biasP 2*1024 f32       =     8,192
//   [3,153,920)  hbuf  2*2*64*256 bf16  =   131,072
//   [3,284,992)  cnt   (padded)         =       256
//   [3,285,248)  xp    2*800*64*1024 bf16 = 209,715,200
// total ~213 MB.
// ---------------------------------------------------------------------------
extern "C" void kernel_launch(void* const* d_in, const int* in_sizes, int n_in,
                              void* d_out, int out_size, void* d_ws,
                              size_t ws_size, hipStream_t stream) {
  const float* x   = (const float*)d_in[0];
  const float* Wxf = (const float*)d_in[1];
  const float* Whf = (const float*)d_in[2];
  const float* bfo = (const float*)d_in[3];
  const float* Wxb = (const float*)d_in[4];
  const float* Whb = (const float*)d_in[5];
  const float* bbo = (const float*)d_in[6];
  float* out = (float*)d_out;

  char* ws = (char*)d_ws;
  __bf16* WxT   = (__bf16*)(ws);
  __bf16* WhT   = (__bf16*)(ws + 2097152);
  float*  biasP = (float*)(ws + 3145728);
  __bf16* hbuf  = (__bf16*)(ws + 3153920);
  int*    cnt   = (int*)(ws + 3284992);
  __bf16* xp    = (__bf16*)(ws + 3285248);

  pack_kernel<<<4096, 256, 0, stream>>>(Wxf, Whf, bfo, Wxb, Whb, bbo,
                                        WxT, WhT, biasP, hbuf, cnt);

  dim3 gx(Tn, 8);  // 800 time tiles x (2 dirs * 4 N-blocks)
  xproj_kernel<<<gx, 256, 64 * 1024, stream>>>(x, WxT, biasP, xp);

  lstm_scan_kernel<<<8, 256, 112 * 1024, stream>>>(xp, WhT, hbuf, cnt, out);
}
